// GRUNet_39178691674958
// MI455X (gfx1250) — matrix operand order repaired and checked by reading, not compiled, not run on hardware
//
#include <hip/hip_runtime.h>
#include <hip/hip_bf16.h>
#include <math.h>
#include <stdint.h>

typedef __attribute__((ext_vector_type(16))) _Float16 v16h;
typedef __attribute__((ext_vector_type(8)))  _Float16 v8h;
typedef __attribute__((ext_vector_type(8)))  float    v8f;
typedef __attribute__((ext_vector_type(4)))  int      v4i;

#define BATCH 256
#define SEQ   256
#define IN_DIM 128
#define HID   512
#define KMAXO 4
#define MID   256
#define NCLS  16

#define LDH   (HID + 8)      // padded LDS row stride (halves/floats) -> bank rotate
#define LDX   (IN_DIM + 8)

#define GAS __attribute__((address_space(1)))
typedef GAS const _Float16 gf16;   // explicit global-AS f16 (forces global_load)

// ---------------------------------------------------------------------------
// Async global->LDS (gfx1250) with safe fallback
// ---------------------------------------------------------------------------
#if __has_builtin(__builtin_amdgcn_global_load_async_to_lds_b128) && \
    __has_builtin(__builtin_amdgcn_s_wait_asynccnt)
#define GRU_HAS_ASYNC 1
#else
#define GRU_HAS_ASYNC 0
#endif

#define AS1P __attribute__((address_space(1))) v4i*
#define AS3P __attribute__((address_space(3))) v4i*

__device__ __forceinline__ void async_copy16(const _Float16* g, _Float16* l) {
#if GRU_HAS_ASYNC
  // global segment address == flat address for global memory;
  // LDS offset == addr[31:0] of the flat LDS address (ISA aperture rule).
  __builtin_amdgcn_global_load_async_to_lds_b128(
      (AS1P)(uintptr_t)g,
      (AS3P)(unsigned int)(uintptr_t)l,
      0, 0);
#else
  *(v8h*)l = *(const v8h*)g;
#endif
}
__device__ __forceinline__ void async_join() {
#if GRU_HAS_ASYNC
  __builtin_amdgcn_s_wait_asynccnt(0);
#endif
}

// ---------------------------------------------------------------------------
// WMMA helpers (wave32, v_wmma_f32_16x16x32_f16)
// A 16x32: lane L: row L&15, elems i = K[(L>>4)*16+i].
// B 32x16 from PRE-TRANSPOSED [N,K]: lane L: n=L&15, elems i = K[(L>>4)*16+i].
// C/D f32: lane L: n=L&15, VGPR v -> row v + 8*(L>>4).
// ---------------------------------------------------------------------------
__device__ __forceinline__ v16h load_A16(const _Float16* base,
                                         int ld, int row0, int k0) {
  const int lane = threadIdx.x & 31;
  const int m  = lane & 15;
  const int kb = (lane >> 4) * 16;
  const _Float16* p = base + (size_t)(row0 + m) * (size_t)ld + k0 + kb;
  v8h lo = *(const v8h*)(p);
  v8h hi = *(const v8h*)(p + 8);
  v16h a;
#pragma unroll
  for (int i = 0; i < 8; ++i) { a[i] = lo[i]; a[i + 8] = hi[i]; }
  return a;
}

__device__ __forceinline__ v16h load_BT16(const _Float16* bt,
                                          int ld, int col0, int k0) {
  const int lane = threadIdx.x & 31;
  const int n  = lane & 15;
  const int kb = (lane >> 4) * 16;
  const _Float16* p = bt + (size_t)(col0 + n) * (size_t)ld + k0 + kb;
  v8h lo = *(const v8h*)(p);
  v8h hi = *(const v8h*)(p + 8);
  v16h b;
#pragma unroll
  for (int i = 0; i < 8; ++i) { b[i] = lo[i]; b[i + 8] = hi[i]; }
  return b;
}

// Address-space(1) variant: guarantees global_load_b128 even when the base
// pointer is opaque to address-space inference (asm-rotated pointers).
__device__ __forceinline__ v16h load_BT16g(gf16* bt, int ld, int col0, int k0) {
  const int lane = threadIdx.x & 31;
  const int n  = lane & 15;
  const int kb = (lane >> 4) * 16;
  gf16* p = bt + (size_t)(col0 + n) * (size_t)ld + k0 + kb;
  v8h lo = *(GAS const v8h*)(p);
  v8h hi = *(GAS const v8h*)(p + 8);
  v16h b;
#pragma unroll
  for (int i = 0; i < 8; ++i) { b[i] = lo[i]; b[i + 8] = hi[i]; }
  return b;
}

__device__ __forceinline__ v8f wmma16(v16h a, v16h b, v8f c) {
  return __builtin_amdgcn_wmma_f32_16x16x32_f16(false, a, false, b,
                                                (short)0, c, false, false);
}

__device__ __forceinline__ float hsig(float x) {
  return fminf(fmaxf(0.2f * x + 0.5f, 0.0f), 1.0f);
}

// ---------------------------------------------------------------------------
// Preprocessing
// ---------------------------------------------------------------------------
__global__ void convert_f16_kernel(const float* __restrict__ in,
                                   _Float16* __restrict__ out, int n) {
  int i = blockIdx.x * blockDim.x + threadIdx.x;
  if (i < n) out[i] = (_Float16)in[i];
}

// in: [R,C] f32 row-major -> out: [C,R] f16 row-major
__global__ void transpose_f16_kernel(const float* __restrict__ in,
                                     _Float16* __restrict__ out, int R, int C) {
  int i = blockIdx.x * blockDim.x + threadIdx.x;
  if (i >= R * C) return;
  int r = i / C, c = i % C;
  out[(size_t)c * R + r] = (_Float16)in[i];
}

// ---------------------------------------------------------------------------
// Persistent GRU: one launch for all 256 steps.
// Row-independent recurrence: block owns 16 batch rows; h lives in LDS.
// 16 blocks x 512 threads (16 waves); wave owns 2 column tiles (32 total).
// Double-buffered async x staging overlaps DMA with phase-2 compute.
// Weight pointers are AS1-typed and rotated through an empty asm each step:
// no LICM register blow-up, and loads stay global_load_b128 (not flat).
// ---------------------------------------------------------------------------
__global__ __launch_bounds__(512) void gru_persistent_kernel(
    const _Float16* __restrict__ x16,                       // [B,S,IN]
    const _Float16* __restrict__ WzT, const _Float16* __restrict__ WrT,
    const _Float16* __restrict__ WhT,                       // [HID,IN]
    const _Float16* __restrict__ UzT, const _Float16* __restrict__ UrT,
    const _Float16* __restrict__ UhT,                       // [HID,HID]
    const float* __restrict__ b_z, const float* __restrict__ b_r,
    const float* __restrict__ b_h,
    _Float16* __restrict__ hout16) {                        // [B,HID]
  __shared__ _Float16 sh_h16[16 * LDH];
  __shared__ float    sh_h32[16 * LDH];
  __shared__ _Float16 sh_rh [16 * LDH];
  __shared__ _Float16 sh_x  [2][16 * LDX];

  const int tid  = threadIdx.x;
  const int wave = tid >> 5;
  const int lane = tid & 31;
  const int row0 = blockIdx.x * 16;     // global batch-row base
  const int ct0  = wave * 2;            // first of this wave's 2 col tiles
  const int nloc = lane & 15;
  const int mh   = 8 * (lane >> 4);
  const int xr   = tid >> 4;            // staging row (tid < 256)
  const int xc   = (tid & 15) * 8;      // staging col (halves)

  // loop-rotated AS1 weight pointers (opaque to LICM, global_load guaranteed)
  gf16 *wzp = (gf16*)(uintptr_t)WzT, *wrp = (gf16*)(uintptr_t)WrT;
  gf16 *whp = (gf16*)(uintptr_t)WhT;
  gf16 *uzp = (gf16*)(uintptr_t)UzT, *urp = (gf16*)(uintptr_t)UrT;
  gf16 *uhp = (gf16*)(uintptr_t)UhT;

  for (int i = tid; i < 16 * LDH; i += 512) {
    sh_h16[i] = (_Float16)0.0f;
    sh_h32[i] = 0.0f;
  }
  // prestage x[:, 0, :]
  if (tid < 256)
    async_copy16(x16 + (size_t)(row0 + xr) * (SEQ * IN_DIM) + xc,
                 &sh_x[0][xr * LDX + xc]);
  async_join();
  __syncthreads();

#pragma unroll 1
  for (int t = 0; t < SEQ; ++t) {
    // break loop-invariance of weight addresses (prevents VGPR spills)
    asm volatile("" : "+v"(wzp), "+v"(wrp), "+v"(whp),
                      "+v"(uzp), "+v"(urp), "+v"(uhp));
    const _Float16* xb = sh_x[t & 1];

    // ---- phase 1: z, r for two column tiles (shared A operand) ----
    v8f az[2] = {}, ar[2] = {};
#pragma unroll
    for (int k0 = 0; k0 < IN_DIM; k0 += 32) {
      v16h a = load_A16(xb, LDX, 0, k0);
#pragma unroll
      for (int u = 0; u < 2; ++u) {
        az[u] = wmma16(a, load_BT16g(wzp, IN_DIM, (ct0 + u) * 16, k0), az[u]);
        ar[u] = wmma16(a, load_BT16g(wrp, IN_DIM, (ct0 + u) * 16, k0), ar[u]);
      }
    }
#pragma unroll 2
    for (int k0 = 0; k0 < HID; k0 += 32) {
      v16h a = load_A16(sh_h16, LDH, 0, k0);
#pragma unroll
      for (int u = 0; u < 2; ++u) {
        az[u] = wmma16(a, load_BT16g(uzp, HID, (ct0 + u) * 16, k0), az[u]);
        ar[u] = wmma16(a, load_BT16g(urp, HID, (ct0 + u) * 16, k0), ar[u]);
      }
    }
    float zs[2][8];
#pragma unroll
    for (int u = 0; u < 2; ++u) {
      const int n = (ct0 + u) * 16 + nloc;
      const float bzv = b_z[n], brv = b_r[n];
#pragma unroll
      for (int v = 0; v < 8; ++v) {
        const int idx = (v + mh) * LDH + n;
        float z = hsig(az[u][v] + bzv);
        float r = hsig(ar[u][v] + brv);
        zs[u][v] = z;
        sh_rh[idx] = (_Float16)(r * sh_h32[idx]);
      }
    }

    // ---- stage x[:, t+1, :] into the other buffer (overlaps phase 2) ----
    if (t + 1 < SEQ && tid < 256)
      async_copy16(x16 + (size_t)(row0 + xr) * (SEQ * IN_DIM)
                       + (t + 1) * IN_DIM + xc,
                   &sh_x[(t + 1) & 1][xr * LDX + xc]);
    __syncthreads();   // r*h visible to all waves

    // ---- phase 2: hh = tanh(x@W_h + (r*h)@U_h + b_h); h = z*h + (1-z)*hh ----
    v8f ah[2] = {};
#pragma unroll
    for (int k0 = 0; k0 < IN_DIM; k0 += 32) {
      v16h a = load_A16(xb, LDX, 0, k0);
#pragma unroll
      for (int u = 0; u < 2; ++u)
        ah[u] = wmma16(a, load_BT16g(whp, IN_DIM, (ct0 + u) * 16, k0), ah[u]);
    }
#pragma unroll 4
    for (int k0 = 0; k0 < HID; k0 += 32) {
      v16h a = load_A16(sh_rh, LDH, 0, k0);
#pragma unroll
      for (int u = 0; u < 2; ++u)
        ah[u] = wmma16(a, load_BT16g(uhp, HID, (ct0 + u) * 16, k0), ah[u]);
    }
#pragma unroll
    for (int u = 0; u < 2; ++u) {
      const int n = (ct0 + u) * 16 + nloc;
      const float bhv = b_h[n];
#pragma unroll
      for (int v = 0; v < 8; ++v) {
        const int idx = (v + mh) * LDH + n;
        float hh = tanhf(ah[u][v] + bhv);
        float z  = zs[u][v];
        float hn = z * sh_h32[idx] + (1.0f - z) * hh;
        sh_h32[idx] = hn;
        sh_h16[idx] = (_Float16)hn;
      }
    }
    async_join();      // next-step x DMA complete (issuing waves)
    __syncthreads();   // h + next x visible to all waves
  }

  // ---- flush final hidden state (f16) for the MLP head ----
  for (int i = tid; i < 16 * HID; i += 512) {
    const int r = i / HID, c = i % HID;
    hout16[(size_t)(row0 + r) * HID + c] = sh_h16[r * LDH + c];
  }
}

// ---------------------------------------------------------------------------
// Maxout: mid = max_k (h @ mo_W[k] + mo_b[k]).  256 tiles of 16x16.
// ---------------------------------------------------------------------------
__global__ __launch_bounds__(256) void maxout_kernel(
    const _Float16* __restrict__ h16,
    const _Float16* __restrict__ moWT,   // [K][MID][HID]
    const float* __restrict__ mo_b,      // [K][MID]
    _Float16* __restrict__ mid16) {
  const int tile = blockIdx.x * 8 + (threadIdx.x >> 5);
  const int rt = tile >> 4, ct = tile & 15;
  const int row0 = rt * 16, col0 = ct * 16;
  const int lane = threadIdx.x & 31;
  const int n = col0 + (lane & 15);

  float best[8];
  for (int kk = 0; kk < KMAXO; ++kk) {
    v8f acc = {};
    const _Float16* bt = moWT + (size_t)kk * MID * HID;
    for (int k0 = 0; k0 < HID; k0 += 32) {
      v16h a = load_A16(h16, HID, row0, k0);
      v16h b = load_BT16(bt, HID, col0, k0);
      acc = wmma16(a, b, acc);
    }
    const float bias = mo_b[kk * MID + n];
#pragma unroll
    for (int v = 0; v < 8; ++v) {
      float val = acc[v] + bias;
      best[v] = (kk == 0) ? val : fmaxf(best[v], val);
    }
  }
#pragma unroll
  for (int v = 0; v < 8; ++v) {
    const int m = row0 + v + 8 * (lane >> 4);
    mid16[(size_t)m * MID + n] = (_Float16)best[v];
  }
}

// ---------------------------------------------------------------------------
// Dense (linear): out = A @ W + b, A:[256,MID], W:[MID,MID]; f32+f16 out
// ---------------------------------------------------------------------------
__global__ __launch_bounds__(256) void linear_kernel(
    const _Float16* __restrict__ A16, const _Float16* __restrict__ WT,
    const float* __restrict__ bias,
    float* __restrict__ out32, _Float16* __restrict__ out16) {
  const int tile = blockIdx.x * 8 + (threadIdx.x >> 5);
  const int rt = tile >> 4, ct = tile & 15;
  const int row0 = rt * 16, col0 = ct * 16;
  v8f acc = {};
  for (int k0 = 0; k0 < MID; k0 += 32) {
    v16h a = load_A16(A16, MID, row0, k0);
    v16h b = load_BT16(WT, MID, col0, k0);
    acc = wmma16(a, b, acc);
  }
  const int lane = threadIdx.x & 31;
  const int n = col0 + (lane & 15);
  const float bv = bias[n];
#pragma unroll
  for (int v = 0; v < 8; ++v) {
    const int m = row0 + v + 8 * (lane >> 4);
    const size_t idx = (size_t)m * MID + n;
    float val = acc[v] + bv;
    out32[idx] = val;
    out16[idx] = (_Float16)val;
  }
}

// ---------------------------------------------------------------------------
// Highway: t = sigmoid(h@Wc+bc); hh = relu(h@W+b); out = t*hh + (1-t)*h
// ---------------------------------------------------------------------------
__global__ __launch_bounds__(256) void highway_kernel(
    const _Float16* __restrict__ h16, const float* __restrict__ h32,
    const _Float16* __restrict__ WcT, const _Float16* __restrict__ WT,
    const float* __restrict__ bc, const float* __restrict__ b,
    _Float16* __restrict__ out16) {
  const int tile = blockIdx.x * 8 + (threadIdx.x >> 5);
  const int rt = tile >> 4, ct = tile & 15;
  const int row0 = rt * 16, col0 = ct * 16;
  v8f at = {}; v8f ah = {};
  for (int k0 = 0; k0 < MID; k0 += 32) {
    v16h a  = load_A16(h16, MID, row0, k0);
    v16h bt = load_BT16(WcT, MID, col0, k0);
    v16h bh = load_BT16(WT, MID, col0, k0);
    at = wmma16(a, bt, at);
    ah = wmma16(a, bh, ah);
  }
  const int lane = threadIdx.x & 31;
  const int n = col0 + (lane & 15);
  const float bcv = bc[n], bv = b[n];
#pragma unroll
  for (int v = 0; v < 8; ++v) {
    const int m = row0 + v + 8 * (lane >> 4);
    const size_t idx = (size_t)m * MID + n;
    float tg = 1.0f / (1.0f + expf(-(at[v] + bcv)));
    float hv = fmaxf(ah[v] + bv, 0.0f);
    out16[idx] = (_Float16)(tg * hv + (1.0f - tg) * h32[idx]);
  }
}

// ---------------------------------------------------------------------------
// Logits: [256, NCLS] = h @ d2_W + d2_b.  16 tiles (one col tile).
// ---------------------------------------------------------------------------
__global__ __launch_bounds__(256) void logits_kernel(
    const _Float16* __restrict__ h16, const _Float16* __restrict__ WT, // [NCLS,MID]
    const float* __restrict__ bias, float* __restrict__ logits) {
  const int tile = blockIdx.x * 8 + (threadIdx.x >> 5); // 0..15
  const int row0 = tile * 16;
  v8f acc = {};
  for (int k0 = 0; k0 < MID; k0 += 32) {
    v16h a = load_A16(h16, MID, row0, k0);
    v16h b = load_BT16(WT, MID, 0, k0);
    acc = wmma16(a, b, acc);
  }
  const int lane = threadIdx.x & 31;
  const int n = lane & 15;
  const float bv = bias[n];
#pragma unroll
  for (int v = 0; v < 8; ++v) {
    const int m = row0 + v + 8 * (lane >> 4);
    logits[(size_t)m * NCLS + n] = acc[v] + bv;
  }
}

__global__ void softmax_kernel(const float* __restrict__ logits,
                               float* __restrict__ out) {
  int r = blockIdx.x * blockDim.x + threadIdx.x;
  if (r >= BATCH) return;
  const float* lp = logits + (size_t)r * NCLS;
  float mx = lp[0];
#pragma unroll
  for (int j = 1; j < NCLS; ++j) mx = fmaxf(mx, lp[j]);
  float s = 0.0f;
  float e[NCLS];
#pragma unroll
  for (int j = 0; j < NCLS; ++j) { e[j] = expf(lp[j] - mx); s += e[j]; }
  float inv = 1.0f / s;
#pragma unroll
  for (int j = 0; j < NCLS; ++j) out[(size_t)r * NCLS + j] = e[j] * inv;
}

// ---------------------------------------------------------------------------
// Host launcher
// ---------------------------------------------------------------------------
extern "C" void kernel_launch(void* const* d_in, const int* in_sizes, int n_in,
                              void* d_out, int out_size, void* d_ws, size_t ws_size,
                              hipStream_t stream) {
  const float* x    = (const float*)d_in[0];
  const float* W_z  = (const float*)d_in[1];
  const float* W_r  = (const float*)d_in[2];
  const float* W_h  = (const float*)d_in[3];
  const float* U_z  = (const float*)d_in[4];
  const float* U_r  = (const float*)d_in[5];
  const float* U_h  = (const float*)d_in[6];
  const float* b_z  = (const float*)d_in[7];
  const float* b_r  = (const float*)d_in[8];
  const float* b_h  = (const float*)d_in[9];
  const float* mo_W = (const float*)d_in[10];
  const float* mo_b = (const float*)d_in[11];
  const float* d1_W = (const float*)d_in[12];
  const float* d1_b = (const float*)d_in[13];
  const float* hw_W = (const float*)d_in[14];
  const float* hw_b = (const float*)d_in[15];
  const float* hw_Wc= (const float*)d_in[16];
  const float* hw_bc= (const float*)d_in[17];
  const float* d2_W = (const float*)d_in[18];
  const float* d2_b = (const float*)d_in[19];

  char* ws = (char*)d_ws;
  size_t off = 0;
  auto take = [&](size_t bytes) -> char* {
    char* p = ws + off;
    off = (off + bytes + 255) & ~(size_t)255;
    return p;
  };

  _Float16* x16   = (_Float16*)take((size_t)BATCH * SEQ * IN_DIM * 2);
  _Float16* WzT   = (_Float16*)take((size_t)HID * IN_DIM * 2);
  _Float16* WrT   = (_Float16*)take((size_t)HID * IN_DIM * 2);
  _Float16* WhT   = (_Float16*)take((size_t)HID * IN_DIM * 2);
  _Float16* UzT   = (_Float16*)take((size_t)HID * HID * 2);
  _Float16* UrT   = (_Float16*)take((size_t)HID * HID * 2);
  _Float16* UhT   = (_Float16*)take((size_t)HID * HID * 2);
  _Float16* moWT  = (_Float16*)take((size_t)KMAXO * MID * HID * 2);
  _Float16* d1WT  = (_Float16*)take((size_t)MID * MID * 2);
  _Float16* hwWT  = (_Float16*)take((size_t)MID * MID * 2);
  _Float16* hwWcT = (_Float16*)take((size_t)MID * MID * 2);
  _Float16* d2WT  = (_Float16*)take((size_t)NCLS * MID * 2);
  _Float16* h16   = (_Float16*)take((size_t)BATCH * HID * 2);
  _Float16* mid16 = (_Float16*)take((size_t)BATCH * MID * 2);
  float*    h1_32 = (float*)take((size_t)BATCH * MID * 4);
  _Float16* h1_16 = (_Float16*)take((size_t)BATCH * MID * 2);
  _Float16* h2_16 = (_Float16*)take((size_t)BATCH * MID * 2);
  float*    logit = (float*)take((size_t)BATCH * NCLS * 4);

  // ---- preprocessing: f16 conversion + weight transposes ----
  {
    int n = BATCH * SEQ * IN_DIM;
    convert_f16_kernel<<<(n + 255) / 256, 256, 0, stream>>>(x, x16, n);
  }
  auto tr = [&](const float* in, _Float16* out, int R, int C) {
    int n = R * C;
    transpose_f16_kernel<<<(n + 255) / 256, 256, 0, stream>>>(in, out, R, C);
  };
  tr(W_z, WzT, IN_DIM, HID);
  tr(W_r, WrT, IN_DIM, HID);
  tr(W_h, WhT, IN_DIM, HID);
  tr(U_z, UzT, HID, HID);
  tr(U_r, UrT, HID, HID);
  tr(U_h, UhT, HID, HID);
  for (int kk = 0; kk < KMAXO; ++kk)
    tr(mo_W + (size_t)kk * HID * MID, moWT + (size_t)kk * MID * HID, HID, MID);
  tr(d1_W, d1WT, MID, MID);
  tr(hw_W, hwWT, MID, MID);
  tr(hw_Wc, hwWcT, MID, MID);
  tr(d2_W, d2WT, MID, NCLS);

  // ---- entire GRU recurrence: ONE launch, h resident in LDS ----
  gru_persistent_kernel<<<16, 512, 0, stream>>>(x16, WzT, WrT, WhT,
                                                UzT, UrT, UhT,
                                                b_z, b_r, b_h, h16);

  // ---- head ----
  maxout_kernel<<<32, 256, 0, stream>>>(h16, moWT, mo_b, mid16);
  linear_kernel<<<32, 256, 0, stream>>>(mid16, d1WT, d1_b, h1_32, h1_16);
  highway_kernel<<<32, 256, 0, stream>>>(h1_16, h1_32, hwWcT, hwWT, hw_bc, hw_b,
                                         h2_16);
  logits_kernel<<<2, 256, 0, stream>>>(h2_16, d2WT, d2_b, logit);
  softmax_kernel<<<1, 256, 0, stream>>>(logit, (float*)d_out);
}
